// QuantizedLinear_23776938950951
// MI455X (gfx1250) — matrix operand order repaired and checked by reading, not compiled, not run on hardware
//
#include <hip/hip_runtime.h>

// ---------------------------------------------------------------------------
// Q4 block-dequant GEMM: Y[M,N] = X[M,K] * dequant(Wq)[N,K]^T
//   M = 8192 (4*2048), N = 4096, K = 4096, quant block = 128 (32 blocks/row)
// Tiered on workspace:
//   ws >= 96MB: X pre-converted f16 + W pre-dequantized f16 -> GEMM main loop
//               is pure async-copy + ds_load + WMMA (no conversion VALU).
//   ws >= 32MB: W pre-dequantized, X converted inline.
//   else      : fully fused (inline dequant + convert).
// GEMM: f16 WMMA (V_WMMA_F32_16X16X32_F16), fp32 accumulate.
// Block tile 128(M) x 256(N), 8 waves (wave32) as 2x4, wave tile 64x64
// (4x4 fragments -> 1.0 ds_load_b128 per WMMA). K-tile 64, ping-pong LDS,
// one s_wait_asynccnt + barrier per K-tile.
// ---------------------------------------------------------------------------

typedef _Float16 v16h __attribute__((ext_vector_type(16)));
typedef _Float16 h2   __attribute__((ext_vector_type(2)));
typedef float    v8f  __attribute__((ext_vector_type(8)));
typedef unsigned int v4u __attribute__((ext_vector_type(4)));
typedef unsigned int v2u __attribute__((ext_vector_type(2)));

#define KDIM    4096
#define NTB     256     // N tile per block
#define MT      128     // M tile per block
#define KT      64      // K tile per iteration
#define KTILES  (KDIM / KT)
#define KP      72      // padded LDS row stride (elements): 144B -> 4-bank lane skew
#define PACKROW 2048    // int32s per weight row (32 blocks * 64)

union U32H { unsigned u; h2 h; };
union ABFrag { v4u u[2]; v16h v; };

// two fp32 -> packed f16 (RTZ) in a single VALU op
__device__ __forceinline__ unsigned cvt2h(float a, float b) {
    union { __decltype(__builtin_amdgcn_cvt_pkrtz(0.f, 0.f)) v; unsigned u; } r;
    r.v = __builtin_amdgcn_cvt_pkrtz(a, b);
    return r.u;
}

// one packed-int -> two dequantized f16 weights (K=2c: hi nibble, K=2c+1: lo)
__device__ __forceinline__ unsigned dq2(int v, h2 s2) {
    const U32H m2 = { 0xE408E408u };  // packed f16 -1032.0
    U32H q;
    q.u = ((((unsigned)v << 16) | ((unsigned)v >> 4)) & 0x000F000Fu) | 0x64006400u;
    U32H d;
    d.h = (q.h + m2.h) * s2;          // v_pk_add_f16 + v_pk_mul_f16
    return d.u;
}

// async 16B global -> LDS copy (ASYNCcnt); lds generic ptr low 32 bits = offset
__device__ __forceinline__ void async_cp16(const unsigned short* gp,
                                           unsigned short* lp) {
    unsigned loff = (unsigned)(unsigned long long)lp;
    asm volatile("global_load_async_to_lds_b128 %0, %1, off"
                 :: "v"(loff), "v"((unsigned long long)gp) : "memory");
}

// ---------------------------------------------------------------------------
// Pre-pass 1: X fp32 -> f16 row-major [M][K] (8 values / thread)
// ---------------------------------------------------------------------------
__global__ __launch_bounds__(256)
void x_convert_f16(const float* __restrict__ X, unsigned short* __restrict__ X16) {
    const long i = (long)blockIdx.x * 256 + threadIdx.x;
    const float4* xp = (const float4*)X + i * 2;
    const float4 a = xp[0], b = xp[1];
    v4u p;
    p.x = cvt2h(a.x, a.y);
    p.y = cvt2h(a.z, a.w);
    p.z = cvt2h(b.x, b.y);
    p.w = cvt2h(b.z, b.w);
    *((v4u*)X16 + i) = p;
}

// ---------------------------------------------------------------------------
// Pre-pass 2: dequantize packed int4 weights to f16 [N][K] row-major.
// ---------------------------------------------------------------------------
__global__ __launch_bounds__(256)
void w_dequant_f16(const int* __restrict__ Wq, const float* __restrict__ S,
                   unsigned short* __restrict__ W16) {
    const long i  = (long)blockIdx.x * 256 + threadIdx.x;
    const long c4 = i * 4;                    // first int32 index
    const int row = (int)(c4 >> 11);          // 2048 ints per weight row
    const int col = (int)(c4 & 2047);
    const int blk = col >> 6;                 // 64 ints per quant block
    const int cb  = col & 63;
    const float s = S[(long)row * 32 + blk];
    U32H s2; s2.u = cvt2h(s, s);
    const int4 w = *(const int4*)(Wq + c4);
    v4u p;
    p.x = dq2(w.x, s2.h);
    p.y = dq2(w.y, s2.h);
    p.z = dq2(w.z, s2.h);
    p.w = dq2(w.w, s2.h);
    *(v4u*)(&W16[(long)row * KDIM + blk * 128 + cb * 2]) = p;
}

// ---------------------------------------------------------------------------
// Main GEMM.
// ---------------------------------------------------------------------------
template <bool PREX, bool PREW>
__global__ __launch_bounds__(256)
void q4gemm_f16_wmma(const float* __restrict__ X,
                     const int*   __restrict__ Wq,
                     const float* __restrict__ S,
                     const unsigned short* __restrict__ X16,
                     const unsigned short* __restrict__ W16,
                     float*       __restrict__ Y,
                     int Ndim) {
    __shared__ unsigned short Xs[2][MT  * KP];  // 2 x 18 KB
    __shared__ unsigned short Ws[2][NTB * KP];  // 2 x 36 KB

    const int tid  = threadIdx.x;
    const int lane = tid & 31;
    const int wave = tid >> 5;      // 0..7
    const int wm   = wave >> 2;     // 0..1 -> M offset wm*64
    const int wn   = wave & 3;      // 0..3 -> N offset wn*64
    const int m0   = blockIdx.y * MT;
    const int n0   = blockIdx.x * NTB;

    v8f acc[4][4];
#pragma unroll
    for (int i = 0; i < 4; ++i)
#pragma unroll
        for (int j = 0; j < 4; ++j)
#pragma unroll
            for (int r = 0; r < 8; ++r) acc[i][j][r] = 0.0f;

    // staging regs (only live in non-async paths)
    float4 xr[8];
    int4   wr[8];
    float  sr[8];

    auto copy_x = [&](int buf, int kt) {
        unsigned short* xs = &Xs[buf][0];
        if constexpr (PREX) {
            const unsigned short* xb = X16 + (long)m0 * KDIM + kt * KT;
#pragma unroll
            for (int i = 0; i < 4; ++i) {      // 1024 x 16B chunks
                int f   = i * 256 + tid;
                int row = f >> 3;              // 8 chunks per 64-elem row
                int kp  = (f & 7) << 3;
                async_cp16(xb + (long)row * KDIM + kp, &xs[row * KP + kp]);
            }
        } else {
            const float* xb = X + (long)m0 * KDIM + kt * KT;
#pragma unroll
            for (int i = 0; i < 8; ++i) {
                int f   = i * 256 + tid;
                int row = f >> 4;              // 16 float4 per row
                int kp  = (f & 15) << 2;
                xr[i] = *(const float4*)(xb + (long)row * KDIM + kp);
            }
#pragma unroll
            for (int i = 0; i < 8; ++i) {
                int f   = i * 256 + tid;
                int row = f >> 4;
                int kp  = (f & 15) << 2;
                v2u p;
                p.x = cvt2h(xr[i].x, xr[i].y);
                p.y = cvt2h(xr[i].z, xr[i].w);
                *(v2u*)(&xs[row * KP + kp]) = p;
            }
        }
    };

    auto copy_w = [&](int buf, int kt) {
        unsigned short* ws = &Ws[buf][0];
        if constexpr (PREW) {
            const unsigned short* wb = W16 + (long)n0 * KDIM + kt * KT;
#pragma unroll
            for (int i = 0; i < 8; ++i) {      // 2048 x 16B chunks
                int f   = i * 256 + tid;
                int row = f >> 3;
                int kp  = (f & 7) << 3;
                async_cp16(wb + (long)row * KDIM + kp, &ws[row * KP + kp]);
            }
        } else {
            const int blk = kt >> 1;
            const int cb  = (kt & 1) << 5;
            const int* wb = Wq + (long)n0 * PACKROW + blk * 64 + cb;
#pragma unroll
            for (int i = 0; i < 8; ++i) {      // 8192 ints
                int ii  = i * 1024 + tid * 4;
                int row = ii >> 5;
                int c   = ii & 31;
                wr[i] = *(const int4*)(wb + (long)row * PACKROW + c);
                sr[i] = S[(long)(n0 + row) * 32 + blk];
            }
#pragma unroll
            for (int i = 0; i < 8; ++i) {
                int ii  = i * 1024 + tid * 4;
                int row = ii >> 5;
                int c   = ii & 31;
                U32H s2; s2.u = cvt2h(sr[i], sr[i]);
                const int4 w = wr[i];
                v4u p;
                p.x = dq2(w.x, s2.h);
                p.y = dq2(w.y, s2.h);
                p.z = dq2(w.z, s2.h);
                p.w = dq2(w.w, s2.h);
                *(v4u*)(&ws[row * KP + c * 2]) = p;
            }
        }
    };

    // prologue: fill buffer 0
    copy_x(0, 0);
    copy_w(0, 0);
    if constexpr (PREX || PREW) asm volatile("s_wait_asynccnt 0x0" ::: "memory");
    __syncthreads();

    for (int kt = 0; kt < KTILES; ++kt) {
        const int cur = kt & 1;
        if (kt + 1 < KTILES) {
            copy_x(cur ^ 1, kt + 1);
            copy_w(cur ^ 1, kt + 1);
        }

        const unsigned short* xs = &Xs[cur][0];
        const unsigned short* ws = &Ws[cur][0];
#pragma unroll
        for (int ks = 0; ks < 2; ++ks) {
            ABFrag a[4], b[4];
            const int arow = wm * 64 + (lane & 15);
            const int ak   = ks * 32 + ((lane >> 4) << 4);
#pragma unroll
            for (int fm = 0; fm < 4; ++fm) {
                const unsigned short* p = &xs[(arow + fm * 16) * KP + ak];
                a[fm].u[0] = *(const v4u*)(p);
                a[fm].u[1] = *(const v4u*)(p + 8);
            }
            const int brow = wn * 64 + (lane & 15);
            const int bk   = ks * 32 + ((lane >> 4) << 4);
#pragma unroll
            for (int fn = 0; fn < 4; ++fn) {
                const unsigned short* p = &ws[(brow + fn * 16) * KP + bk];
                b[fn].u[0] = *(const v4u*)(p);
                b[fn].u[1] = *(const v4u*)(p + 8);
            }
#pragma unroll
            for (int fm = 0; fm < 4; ++fm)
#pragma unroll
                for (int fn = 0; fn < 4; ++fn)
                    acc[fm][fn] = __builtin_amdgcn_wmma_f32_16x16x32_f16(
                        false, a[fm].v, false, b[fn].v,
                        (short)0, acc[fm][fn], false, false);
        }
        if constexpr (PREX || PREW) asm volatile("s_wait_asynccnt 0x0" ::: "memory");
        __syncthreads();   // publish next buffer / release current
    }

    // C/D layout: lanes 0-15 -> M = r, N = lane ; lanes 16-31 -> M = 8+r
#pragma unroll
    for (int fm = 0; fm < 4; ++fm) {
        const int mr = m0 + wm * 64 + fm * 16 + ((lane >> 4) << 3);
#pragma unroll
        for (int fn = 0; fn < 4; ++fn) {
            const int nc = n0 + wn * 64 + fn * 16 + (lane & 15);
            float* yp = Y + (long)mr * Ndim + nc;
#pragma unroll
            for (int r = 0; r < 8; ++r)
                yp[(long)r * Ndim] = acc[fm][fn][r];
        }
    }
}

extern "C" void kernel_launch(void* const* d_in, const int* in_sizes, int n_in,
                              void* d_out, int out_size, void* d_ws, size_t ws_size,
                              hipStream_t stream) {
    const float* X  = (const float*)d_in[0];   // (4, 2048, 4096) fp32
    const int*   Wq = (const int*)d_in[1];     // (4096, 32, 64) int32 (2 nibbles each)
    const float* S  = (const float*)d_in[2];   // (4096, 32, 1) fp32
    float*       Y  = (float*)d_out;           // (4, 2048, 4096) fp32

    const int M    = in_sizes[0] / KDIM;          // 8192
    const int Ndim = in_sizes[1] / (32 * 64);     // 4096

    dim3 grid(Ndim / NTB, M / MT);  // 16 x 64 = 1024 workgroups
    const size_t xbytes = (size_t)M * KDIM * sizeof(unsigned short);     // 64 MB
    const size_t wbytes = (size_t)Ndim * KDIM * sizeof(unsigned short);  // 32 MB

    if (ws_size >= xbytes + wbytes) {
        unsigned short* X16 = (unsigned short*)d_ws;
        unsigned short* W16 = (unsigned short*)((char*)d_ws + xbytes);
        x_convert_f16<<<(int)((long)M * KDIM / 8 / 256), 256, 0, stream>>>(X, X16);
        w_dequant_f16<<<Ndim * 2, 256, 0, stream>>>(Wq, S, W16);
        q4gemm_f16_wmma<true, true><<<grid, 256, 0, stream>>>(X, Wq, S, X16, W16, Y, Ndim);
    } else if (ws_size >= wbytes) {
        unsigned short* W16 = (unsigned short*)d_ws;
        w_dequant_f16<<<Ndim * 2, 256, 0, stream>>>(Wq, S, W16);
        q4gemm_f16_wmma<false, true><<<grid, 256, 0, stream>>>(X, Wq, S, nullptr, W16, Y, Ndim);
    } else {
        q4gemm_f16_wmma<false, false><<<grid, 256, 0, stream>>>(X, Wq, S, nullptr, nullptr, Y, Ndim);
    }
}
